// NautilusYiJing_45500883534072
// MI455X (gfx1250) — matrix operand order repaired and checked by reading, not compiled, not run on hardware
//
#include <hip/hip_runtime.h>
#include <cstdint>
#include <cstddef>

// ---------------- static config (matches reference) ----------------
#define D_MODEL   1024
#define D_EXPERT  512
#define N_EXPERTS 6
#define TOKENS    8192      // B*T = 4*2048
#define TILE_M    32        // tokens per block (two 16-row WMMA tiles per wave)
#define H_STRIDE  520       // 512 + 8 bf16 pad: keeps 16B alignment, breaks bank conflicts

typedef __bf16 v16bf __attribute__((ext_vector_type(16)));
typedef float  v8f   __attribute__((ext_vector_type(8)));
typedef float  v4f   __attribute__((ext_vector_type(4)));
typedef unsigned int v4u __attribute__((ext_vector_type(4)));
typedef unsigned int v2u __attribute__((ext_vector_type(2)));

union FragBF {
  v16bf v;
  v4u   q[2];
};

__device__ __forceinline__ unsigned short f2bf(float f) {
  unsigned int u = __float_as_uint(f);
  u += 0x7FFFu + ((u >> 16) & 1u);   // round-to-nearest-even bf16
  return (unsigned short)(u >> 16);
}

// branch-free tanh on the native transcendental pipe:
// tanh(x) = 1 - 2/(exp(2x)+1); exp2->{0,inf} saturates result to {-1,+1}.
__device__ __forceinline__ float tanh_fast(float x) {
  float e = __builtin_amdgcn_exp2f(x * 2.885390081777927f);  // 2*log2(e)
  return 1.0f - 2.0f * __builtin_amdgcn_rcpf(e + 1.0f);
}
__device__ __forceinline__ float exp_fast(float x) {
  return __builtin_amdgcn_exp2f(x * 1.4426950408889634f);
}
__device__ __forceinline__ float gelu_tanh(float v) {
  const float k0 = 0.7978845608028654f;   // sqrt(2/pi)
  const float k1 = 0.044715f;
  return 0.5f * v * (1.0f + tanh_fast(k0 * (v + k1 * v * v * v)));
}

// ---------------- router: top-2 softmax over hexagram-anchor logits ----------------
// one wave32 per token
__global__ void __launch_bounds__(256)
router_kernel(const float* __restrict__ x,
              const float* __restrict__ proj_w,
              const float* __restrict__ anchors,
              const float* __restrict__ temp_p,
              float* __restrict__ wts) {
  int token = blockIdx.x * 8 + (threadIdx.x >> 5);
  int lane  = threadIdx.x & 31;
  if (token >= TOKENS) return;
  const float* xr = x + (size_t)token * D_MODEL;

  float z[6] = {0.f, 0.f, 0.f, 0.f, 0.f, 0.f};
  for (int d = lane; d < D_MODEL; d += 32) {
    float xv = xr[d];
#pragma unroll
    for (int e = 0; e < 6; ++e) z[e] += xv * proj_w[e * D_MODEL + d];
  }
#pragma unroll
  for (int off = 16; off > 0; off >>= 1) {
#pragma unroll
    for (int e = 0; e < 6; ++e) z[e] += __shfl_xor(z[e], off, 32);
  }

  if (lane == 0) {
    float q[6], lg[6];
#pragma unroll
    for (int e = 0; e < 6; ++e) q[e] = tanh_fast(z[e] * (1.0f / 0.3f));
    float temp = temp_p[0];
    temp = temp < 0.1f ? 0.1f : temp;
#pragma unroll
    for (int e = 0; e < 6; ++e) {
      float dot = 0.f;
#pragma unroll
      for (int j = 0; j < 6; ++j) dot += q[j] * anchors[e * 6 + j];
      lg[e] = -((6.0f - dot) * 0.5f) / temp;
    }
    // top-2, lowest-index tie-break (matches lax.top_k)
    int i1 = 0;
#pragma unroll
    for (int e = 1; e < 6; ++e) if (lg[e] > lg[i1]) i1 = e;
    int i2 = (i1 == 0) ? 1 : 0;
#pragma unroll
    for (int e = 0; e < 6; ++e) if (e != i1 && lg[e] > lg[i2]) i2 = e;
    float e2 = exp_fast(lg[i2] - lg[i1]);
    float inv = __builtin_amdgcn_rcpf(1.0f + e2);
#pragma unroll
    for (int e = 0; e < 6; ++e) {
      float w = (e == i1) ? inv : ((e == i2) ? e2 * inv : 0.0f);
      wts[(size_t)token * N_EXPERTS + e] = w;
    }
  }
}

// ---------------- x cast: fp32 -> bf16, row-major (A-fragment source) ----------------
__global__ void __launch_bounds__(256)
cast_x_kernel(const float* __restrict__ x, unsigned short* __restrict__ xb) {
  int idx = blockIdx.x * 256 + threadIdx.x;   // float4 index
  v4f v = ((const v4f*)x)[idx];
  v2u p;
  p.x = (unsigned int)f2bf(v.x) | ((unsigned int)f2bf(v.y) << 16);
  p.y = (unsigned int)f2bf(v.z) | ((unsigned int)f2bf(v.w) << 16);
  *(v2u*)(xb + (size_t)idx * 4) = p;
}

// ---------------- weight pack: fp32 [E][R][C] -> bf16 [E][C][R] (K-contiguous) ----------------
__global__ void __launch_bounds__(256)
cast_transpose_kernel(const float* __restrict__ w,
                      unsigned short* __restrict__ wt,
                      int R, int C) {
  long long idx   = (long long)blockIdx.x * blockDim.x + threadIdx.x;
  long long total = (long long)N_EXPERTS * R * C;
  if (idx >= total) return;
  long long rc  = (long long)R * C;
  int e   = (int)(idx / rc);
  int rem = (int)(idx % rc);
  int n = rem / R;
  int k = rem % R;
  wt[idx] = f2bf(w[((size_t)e * R + k) * C + n]);
}

// ---------------- fused MoE: gelu(x@w1)*w_e @ w2, accumulated over experts ----------------
// 256 threads = 8 waves; block owns 32 tokens; each wave owns BOTH 16-row M-tiles for
// its N-range, so every B-fragment load feeds two WMMAs (halves L2 weight traffic).
__global__ void __launch_bounds__(256)
moe_kernel(const unsigned short* __restrict__ xb,    // bf16 [8192][1024]
           const float* __restrict__ rwts,           // [8192][6]
           const unsigned short* __restrict__ w1t,   // [E][512][1024] bf16 (n-major, K contig)
           const unsigned short* __restrict__ w2t,   // [E][1024][512] bf16 (n-major, K contig)
           float* __restrict__ out) {
  __shared__ __align__(16) unsigned short sh_h[TILE_M * H_STRIDE];  // ~33 KB bf16 h-tile
  __shared__ float sh_w[TILE_M * N_EXPERTS];

  const int tid  = threadIdx.x;
  const int wave = tid >> 5;
  const int lane = tid & 31;
  const int t0   = blockIdx.x * TILE_M;

  const int mrow  = lane & 15;
  const int msel  = (lane < 16) ? 0 : 8;    // C/D row-half select
  const int aksel = (lane < 16) ? 0 : 8;    // A-frag K chunk select (16-bit A layout)
  const int bksel = (lane < 16) ? 0 : 8;    // B-frag K chunk select (same 16-bit layout as A)

  if (tid < TILE_M * N_EXPERTS) sh_w[tid] = rwts[(size_t)t0 * N_EXPERTS + tid];
  __syncthreads();

  v8f oacc[2][8] = {};   // 2 M-tiles x 8 N-tiles of fp32 accumulators (persist over experts)

  const unsigned short* xrow0 = xb + (size_t)(t0 + mrow) * D_MODEL;
  const unsigned short* xrow1 = xb + (size_t)(t0 + 16 + mrow) * D_MODEL;

  for (int e = 0; e < N_EXPERTS; ++e) {
    // ---- GEMM1: c1 = x_tile(32x1024) @ w1[e] ; wave owns h columns [wave*64, +64) ----
    v8f c1[2][4] = {};
    const unsigned short* w1e = w1t + (size_t)e * D_EXPERT * D_MODEL;
    for (int k0 = 0; k0 < D_MODEL; k0 += 32) {
      FragBF a0, a1;
      a0.q[0] = *(const v4u*)(xrow0 + k0 + aksel);
      a0.q[1] = *(const v4u*)(xrow0 + k0 + aksel + 16);
      a1.q[0] = *(const v4u*)(xrow1 + k0 + aksel);
      a1.q[1] = *(const v4u*)(xrow1 + k0 + aksel + 16);
      __builtin_prefetch(w1e + (size_t)(wave * 64 + mrow) * D_MODEL +
                             ((k0 + 256) & (D_MODEL - 1)) + bksel, 0, 1);
#pragma unroll
      for (int j = 0; j < 4; ++j) {
        int n = wave * 64 + j * 16 + mrow;
        const unsigned short* bp = w1e + (size_t)n * D_MODEL + k0 + bksel;
        FragBF b;
        b.q[0] = *(const v4u*)bp;
        b.q[1] = *(const v4u*)(bp + 16);
        c1[0][j] = __builtin_amdgcn_wmma_f32_16x16x32_bf16(
            false, a0.v, false, b.v, (short)0, c1[0][j], false, false);
        c1[1][j] = __builtin_amdgcn_wmma_f32_16x16x32_bf16(
            false, a1.v, false, b.v, (short)0, c1[1][j], false, false);
      }
    }

    __syncthreads();   // previous expert's GEMM2 reads of sh_h are done
    // ---- GELU + fold router weight into h rows, write bf16 h-tile ----
#pragma unroll
    for (int mt = 0; mt < 2; ++mt) {
#pragma unroll
      for (int j = 0; j < 4; ++j) {
        int ncol = wave * 64 + j * 16 + mrow;
#pragma unroll
        for (int i = 0; i < 8; ++i) {
          int m = mt * 16 + i + msel;
          float hv = gelu_tanh(c1[mt][j][i]) * sh_w[m * N_EXPERTS + e];
          sh_h[m * H_STRIDE + ncol] = f2bf(hv);
        }
      }
    }
    __syncthreads();   // h-tile ready

    // ---- GEMM2: oacc += h_tile(32x512) @ w2[e] ; wave owns out columns [wave*128, +128) ----
    const unsigned short* w2e = w2t + (size_t)e * D_MODEL * D_EXPERT;
    for (int k0 = 0; k0 < D_EXPERT; k0 += 32) {
      FragBF a0, a1;
      a0.q[0] = *(const v4u*)&sh_h[mrow * H_STRIDE + k0 + aksel];
      a0.q[1] = *(const v4u*)&sh_h[mrow * H_STRIDE + k0 + aksel + 16];
      a1.q[0] = *(const v4u*)&sh_h[(16 + mrow) * H_STRIDE + k0 + aksel];
      a1.q[1] = *(const v4u*)&sh_h[(16 + mrow) * H_STRIDE + k0 + aksel + 16];
      __builtin_prefetch(w2e + (size_t)(wave * 128 + mrow) * D_EXPERT +
                             ((k0 + 256) & (D_EXPERT - 1)) + bksel, 0, 1);
#pragma unroll
      for (int nt = 0; nt < 8; ++nt) {
        int n = wave * 128 + nt * 16 + mrow;
        const unsigned short* bp = w2e + (size_t)n * D_EXPERT + k0 + bksel;
        FragBF b;
        b.q[0] = *(const v4u*)bp;
        b.q[1] = *(const v4u*)(bp + 16);
        oacc[0][nt] = __builtin_amdgcn_wmma_f32_16x16x32_bf16(
            false, a0.v, false, b.v, (short)0, oacc[0][nt], false, false);
        oacc[1][nt] = __builtin_amdgcn_wmma_f32_16x16x32_bf16(
            false, a1.v, false, b.v, (short)0, oacc[1][nt], false, false);
      }
    }
  }

  // ---- store fp32 output tile ----
#pragma unroll
  for (int mt = 0; mt < 2; ++mt) {
#pragma unroll
    for (int nt = 0; nt < 8; ++nt) {
      int col = wave * 128 + nt * 16 + mrow;
#pragma unroll
      for (int i = 0; i < 8; ++i) {
        int m = mt * 16 + i + msel;
        out[(size_t)(t0 + m) * D_MODEL + col] = oacc[mt][nt][i];
      }
    }
  }
}

// ---------------- host-side launcher ----------------
extern "C" void kernel_launch(void* const* d_in, const int* in_sizes, int n_in,
                              void* d_out, int out_size, void* d_ws, size_t ws_size,
                              hipStream_t stream) {
  (void)in_sizes; (void)n_in; (void)out_size; (void)ws_size;

  const float* x       = (const float*)d_in[0];  // (4,2048,1024)
  const float* proj_w  = (const float*)d_in[1];  // (6,1024)
  const float* anchors = (const float*)d_in[2];  // (6,6)
  const float* temp_p  = (const float*)d_in[3];  // scalar
  const float* w1      = (const float*)d_in[4];  // (6,1024,512)
  const float* w2      = (const float*)d_in[5];  // (6,512,1024)
  float* out           = (float*)d_out;          // (4,2048,1024)

  // workspace layout (256-aligned offsets)
  char* ws = (char*)d_ws;
  float*          d_router = (float*)ws;                      // 8192*6*4        = 196608 B
  unsigned short* d_xb  = (unsigned short*)(ws + 196608);     // 8192*1024*2     = 16777216 B
  unsigned short* d_w1t = (unsigned short*)(ws + 196608 + 16777216);            // 6291456 B
  unsigned short* d_w2t = (unsigned short*)(ws + 196608 + 16777216 + 6291456);  // 6291456 B

  // 1) router -> dense top-2 weights
  router_kernel<<<TOKENS / 8, 256, 0, stream>>>(x, proj_w, anchors, temp_p, d_router);

  // 2) cast activations, pack weights bf16 (K-contiguous per output column)
  cast_x_kernel<<<(TOKENS * D_MODEL / 4) / 256, 256, 0, stream>>>(x, d_xb);
  const long long wtotal = (long long)N_EXPERTS * D_MODEL * D_EXPERT;  // 3,145,728
  const int wblocks = (int)((wtotal + 255) / 256);
  cast_transpose_kernel<<<wblocks, 256, 0, stream>>>(w1, d_w1t, D_MODEL, D_EXPERT);
  cast_transpose_kernel<<<wblocks, 256, 0, stream>>>(w2, d_w2t, D_EXPERT, D_MODEL);

  // 3) fused MoE over 32-token tiles
  moe_kernel<<<TOKENS / TILE_M, 256, 0, stream>>>(d_xb, d_router, d_w1t, d_w2t, out);
}
